// RNN_22402549416696
// MI455X (gfx1250) — matrix operand order repaired and checked by reading, not compiled, not run on hardware
//
#include <hip/hip_runtime.h>
#include <hip/hip_bf16.h>
#include <math.h>

// ---------- model dims ----------
#define BB   256
#define SS   256
#define VV   128
#define EE   512
#define HH   512
#define G3   1536      // 3*H
#define DD1  1024
#define DD2  512
#define NTT  12
#define NEE  5
#define MM   (BB*SS)   // 65536
#define BN_EPS 1e-5f

typedef __attribute__((ext_vector_type(16))) _Float16 v16h;
typedef __attribute__((ext_vector_type(8)))  _Float16 v8h;
typedef __attribute__((ext_vector_type(8)))  float    v8f;
typedef __attribute__((ext_vector_type(4)))  unsigned int u32x4;
typedef __attribute__((ext_vector_type(8)))  int      i32x8;
typedef __attribute__((ext_vector_type(4)))  int      i32x4;

// Tensor Data Mover availability (device pass only; host pass just needs signatures)
#if defined(__HIP_DEVICE_COMPILE__) && __has_builtin(__builtin_amdgcn_tensor_load_to_lds) && __has_builtin(__builtin_amdgcn_s_wait_tensorcnt)
#define USE_TDM 1
#else
#define USE_TDM 0
#endif

// scan-kernel LDS layout (bytes)
#define SCAN_LDS_H    (16 * HH * 2)            // 16 KB : h  [16][512] f16
#define SCAN_LDS_HH   (16 * G3 * 4)            // 96 KB : hh [16][1536] f32
#define SCAN_LDS_XI   (16 * G3 * 2)            // 48 KB : xi tile [16][1536] f16
#define SCAN_LDS_TOTAL (SCAN_LDS_H + SCAN_LDS_HH + 2 * SCAN_LDS_XI)   // 208 KB

// ---------------------------------------------------------------------------
// WMMA operand loaders (CDNA5 wave32 16x16x32 f16 layouts, CDNA5 ISA)
// ---------------------------------------------------------------------------
__device__ __forceinline__ v16h load_a16(const _Float16* base /* + m0*ld + k0 */,
                                         int ld, int lane) {
  const int row  = lane & 15;
  const int half = lane >> 4;
  const _Float16* p = base + (size_t)row * ld + half * 8;
  v16h a;
  ((v8h*)&a)[0] = *(const v8h*)(p);
  ((v8h*)&a)[1] = *(const v8h*)(p + 16);
  return a;
}

__device__ __forceinline__ v16h load_b16(const _Float16* base /* + n0*ld + k0 */,
                                         int ld, int lane) {
  const int n    = lane & 15;
  const int half = lane >> 4;
  const _Float16* p = base + (size_t)n * ld + half * 8;
  v16h b;
  ((v8h*)&b)[0] = *(const v8h*)(p);
  ((v8h*)&b)[1] = *(const v8h*)(p + 16);
  return b;
}

__device__ __forceinline__ v8f wmma_f16(v16h a, v16h b, v8f c) {
  return __builtin_amdgcn_wmma_f32_16x16x32_f16(false, a, false, b,
                                                (short)0, c, false, false);
}

// Opaque zero offset: added to a pointer inside a loop it blocks LICM (the
// volatile asm re-defines it each iteration) WITHOUT destroying pointer
// provenance, so the loads stay global_load_b128 (LOADcnt), not flat_load.
__device__ __forceinline__ size_t opaque_zero() {
  unsigned long long z = 0;
  asm volatile("" : "+s"(z));
  return (size_t)z;
}

#if USE_TDM
// ---------------------------------------------------------------------------
// TDM: async-load a [16 rows x G3 f16] tile (row stride SS*G3 elems) into LDS.
// D# per CDNA5 ISA: group0 = flags/lds/global addr,
// group1 = dims/strides. 2D tile, data_size=2B. Groups 2/3 unused (zeros).
// ---------------------------------------------------------------------------
__device__ __forceinline__ void tdm_load_xi_tile(const _Float16* gsrc,
                                                 unsigned lds_byte_off) {
  const unsigned long long ga = (unsigned long long)gsrc;
  const u32x4 g0 = {
      1u,                                             // count=1, user mode
      lds_byte_off,                                   // lds_addr
      (unsigned)ga,                                   // global_addr[31:0]
      ((unsigned)(ga >> 32) & 0x1FFFFFFu) | (2u << 30) // addr[56:32] | type=2
  };
  const unsigned long long rstride = (unsigned long long)SS * G3; // elems
  const i32x8 g1 = {
      (int)(1u << 16),                   // wg_mask=0 | data_size=1 (2B)
      (int)((G3 & 0xFFFFu) << 16),       // abar=0 | tensor_dim0 lo16
      (int)(16u << 16),                  // tensor_dim0 hi16=0 | tensor_dim1 lo16=16
      (int)((unsigned)G3 << 16),         // tensor_dim1 hi=0 | tile_dim0=G3
      (int)16,                           // tile_dim1=16 | tile_dim2=0
      (int)(rstride & 0xFFFFFFFFu),      // tensor_dim0_stride lo32
      (int)((rstride >> 32) & 0xFFFFu),  // stride hi16 | dim1_stride lo16=0
      0                                  // dim1_stride hi32
  };
  const i32x4 z = {0, 0, 0, 0};
#if __clang_major__ >= 23
  const i32x8 z8 = {0, 0, 0, 0, 0, 0, 0, 0};
  __builtin_amdgcn_tensor_load_to_lds(g0, g1, z, z, z8, 0);
#else
  __builtin_amdgcn_tensor_load_to_lds(g0, g1, z, z, 0);
#endif
}
#endif // USE_TDM

// ---------------------------------------------------------------------------
// f32 -> f16 conversion
// ---------------------------------------------------------------------------
__global__ __launch_bounds__(256)
void f32_to_f16_kernel(const float* __restrict__ src, _Float16* __restrict__ dst, int n) {
  int i = blockIdx.x * 256 + threadIdx.x;
  if (i < n) dst[i] = (_Float16)src[i];
}

// ---------------------------------------------------------------------------
// Ensemble mean of We / be
// ---------------------------------------------------------------------------
__global__ __launch_bounds__(256)
void ens_avg_kernel(const float* __restrict__ We, const float* __restrict__ be,
                    float* __restrict__ Wavg, float* __restrict__ bavg) {
  int i = blockIdx.x * 256 + threadIdx.x;
  if (i < NTT * DD2) {
    float s = 0.f;
    #pragma unroll
    for (int e = 0; e < NEE; ++e) s += We[(size_t)e * NTT * DD2 + i];
    Wavg[i] = s * (1.0f / NEE);
  }
  if (i < NTT) {
    float s = 0.f;
    #pragma unroll
    for (int e = 0; e < NEE; ++e) s += be[e * NTT + i];
    bavg[i] = s * (1.0f / NEE);
  }
}

// ---------------------------------------------------------------------------
// WMMA GEMM:  out[M,N] = A[M,K] @ W[N,K]^T + bias[N]
// ---------------------------------------------------------------------------
template<bool GATHER, bool OUT_F16>
__global__ __launch_bounds__(256)
void gemm_wmma_kernel(const _Float16* __restrict__ A,
                      const int*      __restrict__ tok,
                      const _Float16* __restrict__ embh,
                      const _Float16* __restrict__ W,
                      const float*    __restrict__ bias,
                      void* __restrict__ out,
                      int M, int N, int K) {
  const int lane = threadIdx.x & 31;
  const int wave = threadIdx.x >> 5;
  const int wrow = wave >> 2;              // 0..1
  const int wcol = wave & 3;               // 0..3
  const int m0   = blockIdx.y * 32 + wrow * 16;
  const int n0   = blockIdx.x * 256 + wcol * 64;

  v8f acc[4];
  #pragma unroll
  for (int j = 0; j < 4; ++j) acc[j] = (v8f)(0.0f);

  const int row  = lane & 15;
  const int half = lane >> 4;
  const _Float16* arow;
  if (GATHER) {
    const int t = tok[m0 + row];
    arow = embh + (size_t)t * K;
  } else {
    arow = A + (size_t)(m0 + row) * K;
  }

  for (int k0 = 0; k0 < K; k0 += 32) {
    v16h a;
    ((v8h*)&a)[0] = *(const v8h*)(arow + k0 + half * 8);
    ((v8h*)&a)[1] = *(const v8h*)(arow + k0 + 16 + half * 8);

    if (k0 + 32 < K)
      __builtin_prefetch(W + (size_t)(n0 + lane) * K + k0 + 32, 0, 1);

    #pragma unroll
    for (int j = 0; j < 4; ++j) {
      v16h b = load_b16(W + (size_t)(n0 + j * 16) * K + k0, K, lane);
      acc[j] = wmma_f16(a, b, acc[j]);
    }
  }

  #pragma unroll
  for (int j = 0; j < 4; ++j) {
    const int col = n0 + j * 16 + (lane & 15);
    const float bv = bias[col];
    #pragma unroll
    for (int v = 0; v < 8; ++v) {
      const int r = m0 + (lane >> 4) * 8 + v;
      const float val = acc[j][v] + bv;
      if (OUT_F16) ((_Float16*)out)[(size_t)r * N + col] = (_Float16)val;
      else         ((float*)out)   [(size_t)r * N + col] = val;
    }
  }
}

// ---------------------------------------------------------------------------
// GRU recurrence. 16 blocks x 512 thr (16 waves); block owns 16 batch rows.
// h in LDS (WMMA A). Per step: hh = h @ Whh^T (wave: 6 of 96 N-tiles, K=512,
// B streamed from L2 as global_load — opaque zero-offset stops LICM/spilling
// without breaking address-space inference), gate math, and a TDM
// tensor_load_to_lds double-buffered prefetch of the next xi tile.
// ---------------------------------------------------------------------------
__global__ __launch_bounds__(512)
void gru_scan_kernel(const _Float16* __restrict__ xi,   // [M, 1536] f16
                     const _Float16* __restrict__ Whh,  // [1536, 512] f16
                     const float*    __restrict__ bhh,  // [1536]
                     _Float16* __restrict__ seq_out,    // [M, 512] f16 (or null)
                     _Float16* __restrict__ hT_out,     // [B, 512] f16 (or null)
                     int has_seq) {
  extern __shared__ char smem[];
  _Float16* h_sh  = (_Float16*)smem;                       // [16][512]
  float*    hh_sh = (float*)(smem + SCAN_LDS_H);           // [16][1536]
#if USE_TDM
  _Float16* xi_sh0 = (_Float16*)(smem + SCAN_LDS_H + SCAN_LDS_HH);
  _Float16* xi_sh1 = (_Float16*)(smem + SCAN_LDS_H + SCAN_LDS_HH + SCAN_LDS_XI);
#endif

  const int tid  = threadIdx.x;
  const int lane = tid & 31;
  const int wave = tid >> 5;                          // 0..15
  const int bid  = blockIdx.x;                        // 0..15
  const size_t xi_blk = (size_t)bid * 16 * SS * G3;   // elems, start of block's xi

  // h0 = 0
  #pragma unroll
  for (int i = 0; i < 16; ++i) h_sh[i * 512 + tid] = (_Float16)0.0f;
  __syncthreads();

#if USE_TDM
  if (wave == 0)   // one DMA per block: xi tile for t=0 -> buf0
    tdm_load_xi_tile(xi + xi_blk, SCAN_LDS_H + SCAN_LDS_HH);
#endif

  for (int t = 0; t < SS; ++t) {
#if USE_TDM
    if (wave == 0 && t + 1 < SS)   // prefetch next step's xi tile
      tdm_load_xi_tile(xi + xi_blk + (size_t)(t + 1) * G3,
                       SCAN_LDS_H + SCAN_LDS_HH + ((t + 1) & 1) * SCAN_LDS_XI);
#endif

    // ---- hh = h @ Whh^T ----
    v8f acc[6];
    #pragma unroll
    for (int j = 0; j < 6; ++j) acc[j] = (v8f)(0.0f);

    const _Float16* Wt = Whh + opaque_zero();  // per-step: no LICM, stays global
    #pragma unroll 1
    for (int k0 = 0; k0 < HH; k0 += 32) {
      v16h a = load_a16(h_sh + k0, HH, lane);
      #pragma unroll
      for (int j = 0; j < 6; ++j) {
        const int n0 = (wave * 6 + j) * 16;
        v16h b = load_b16(Wt + (size_t)n0 * HH + k0, HH, lane);
        acc[j] = wmma_f16(a, b, acc[j]);
      }
    }
    #pragma unroll
    for (int j = 0; j < 6; ++j) {
      const int col = (wave * 6 + j) * 16 + (lane & 15);
      #pragma unroll
      for (int v = 0; v < 8; ++v)
        hh_sh[((lane >> 4) * 8 + v) * G3 + col] = acc[j][v];
    }

#if USE_TDM
    if (wave == 0) {               // xi[t] tile must be resident; t+1 may fly
      if (t + 1 < SS) __builtin_amdgcn_s_wait_tensorcnt(1);
      else            __builtin_amdgcn_s_wait_tensorcnt(0);
    }
#endif
    __syncthreads();

    // ---- gates: thread owns column j = tid, all 16 rows ----
    const int j = tid;  // 0..511
    const float bhr = bhh[j], bhz = bhh[HH + j], bhn = bhh[2 * HH + j];
#if USE_TDM
    const _Float16* xtile = (t & 1) ? xi_sh1 : xi_sh0;
#endif
    #pragma unroll 4
    for (int rrow = 0; rrow < 16; ++rrow) {
      const int bglob = bid * 16 + rrow;
      const size_t m  = (size_t)bglob * SS + t;
#if USE_TDM
      const _Float16* xrow = xtile + rrow * G3;
#else
      const _Float16* xrow = xi + m * G3;
#endif
      const float xr = (float)xrow[j];
      const float xz = (float)xrow[HH + j];
      const float xn = (float)xrow[2 * HH + j];
      const float hr = hh_sh[rrow * G3 + j] + bhr;
      const float hz = hh_sh[rrow * G3 + HH + j] + bhz;
      const float hn = hh_sh[rrow * G3 + 2 * HH + j] + bhn;
      const float r  = 1.0f / (1.0f + __expf(-(xr + hr)));
      const float z  = 1.0f / (1.0f + __expf(-(xz + hz)));
      const float n  = tanhf(xn + r * hn);
      const float hp = (float)h_sh[rrow * 512 + j];
      const float hv = (1.0f - z) * n + z * hp;
      h_sh[rrow * 512 + j] = (_Float16)hv;
      if (has_seq)  seq_out[m * HH + j] = (_Float16)hv;
      if (!has_seq && t == SS - 1) hT_out[(size_t)bglob * HH + j] = (_Float16)hv;
    }
    __syncthreads();
  }
}

// ---------------------------------------------------------------------------
// BatchNorm (batch stats, ddof=0) + ReLU, f32 in -> f16 out.
// ---------------------------------------------------------------------------
__global__ __launch_bounds__(256)
void bn_relu_kernel(const float* __restrict__ X, const float* __restrict__ g,
                    const float* __restrict__ beta, _Float16* __restrict__ Y,
                    int rows, int N) {
  const int c = blockIdx.x * 256 + threadIdx.x;
  if (c >= N) return;
  float s = 0.f, s2 = 0.f;
  for (int r = 0; r < rows; ++r) {
    const float x = X[(size_t)r * N + c];
    s += x; s2 += x * x;
  }
  const float mean = s / rows;
  const float var  = s2 / rows - mean * mean;
  const float rs   = rsqrtf(var + BN_EPS);
  const float gc = g[c], bc = beta[c];
  for (int r = 0; r < rows; ++r) {
    float y = gc * (X[(size_t)r * N + c] - mean) * rs + bc;
    Y[(size_t)r * N + c] = (_Float16)fmaxf(y, 0.0f);
  }
}

// ---------------------------------------------------------------------------
// Scorer: out[b,t] = sigmoid( h2[b,:] . Wavg[t,:] + bavg[t] )
// ---------------------------------------------------------------------------
__global__ __launch_bounds__(256)
void score_kernel(const _Float16* __restrict__ h2, const float* __restrict__ Wavg,
                  const float* __restrict__ bavg, float* __restrict__ out) {
  const int b = threadIdx.x;     // 0..255
  const int t = blockIdx.x;      // 0..11
  float acc = 0.f;
  const _Float16* hr = h2 + (size_t)b * DD2;
  const float* wr = Wavg + (size_t)t * DD2;
  #pragma unroll 8
  for (int d = 0; d < DD2; ++d) acc += (float)hr[d] * wr[d];
  acc += bavg[t];
  out[b * NTT + t] = 1.0f / (1.0f + __expf(-acc));
}

// ---------------------------------------------------------------------------
extern "C" void kernel_launch(void* const* d_in, const int* in_sizes, int n_in,
                              void* d_out, int out_size, void* d_ws, size_t ws_size,
                              hipStream_t stream) {
  const int*   smiles = (const int*)  d_in[0];
  const float* emb    = (const float*)d_in[1];
  const float* W_ih0  = (const float*)d_in[2];
  const float* W_hh0  = (const float*)d_in[3];
  const float* b_ih0  = (const float*)d_in[4];
  const float* b_hh0  = (const float*)d_in[5];
  const float* W_ih1  = (const float*)d_in[6];
  const float* W_hh1  = (const float*)d_in[7];
  const float* b_ih1  = (const float*)d_in[8];
  const float* b_hh1  = (const float*)d_in[9];
  const float* Wd1    = (const float*)d_in[10];
  const float* bd1    = (const float*)d_in[11];
  const float* g1     = (const float*)d_in[12];
  const float* beta1  = (const float*)d_in[13];
  const float* Wd2    = (const float*)d_in[14];
  const float* bd2    = (const float*)d_in[15];
  const float* g2     = (const float*)d_in[16];
  const float* beta2  = (const float*)d_in[17];
  const float* We     = (const float*)d_in[18];
  const float* be     = (const float*)d_in[19];
  float* out = (float*)d_out;

  // ---- workspace carve-out ----
  char* ws = (char*)d_ws;
  size_t cur = 0;
  auto alloc = [&](size_t bytes) -> char* {
    char* p = ws + cur;
    cur = (cur + bytes + 255) & ~(size_t)255;
    return p;
  };
  _Float16* xi     = (_Float16*)alloc((size_t)MM * G3 * 2);   // 192 MB, reused
  _Float16* seq0   = (_Float16*)alloc((size_t)MM * HH * 2);   // 64 MB
  _Float16* embh   = (_Float16*)alloc((size_t)VV * EE * 2);
  _Float16* Wih0h  = (_Float16*)alloc((size_t)G3 * EE * 2);
  _Float16* Whh0h  = (_Float16*)alloc((size_t)G3 * HH * 2);
  _Float16* Wih1h  = (_Float16*)alloc((size_t)G3 * HH * 2);
  _Float16* Whh1h  = (_Float16*)alloc((size_t)G3 * HH * 2);
  _Float16* Wd1h   = (_Float16*)alloc((size_t)DD1 * HH * 2);
  _Float16* Wd2h   = (_Float16*)alloc((size_t)DD2 * DD1 * 2);
  _Float16* hTh    = (_Float16*)alloc((size_t)BB * HH * 2);
  float*    d1out  = (float*)   alloc((size_t)BB * DD1 * 4);
  _Float16* h1h    = (_Float16*)alloc((size_t)BB * DD1 * 2);
  float*    d2out  = (float*)   alloc((size_t)BB * DD2 * 4);
  _Float16* h2h    = (_Float16*)alloc((size_t)BB * DD2 * 2);
  float*    Wavg   = (float*)   alloc((size_t)NTT * DD2 * 4);
  float*    bavg   = (float*)   alloc((size_t)NTT * 4);

  auto cvt = [&](const float* s, _Float16* d, int n) {
    f32_to_f16_kernel<<<(n + 255) / 256, 256, 0, stream>>>(s, d, n);
  };
  cvt(emb,   embh,  VV * EE);
  cvt(W_ih0, Wih0h, G3 * EE);
  cvt(W_hh0, Whh0h, G3 * HH);
  cvt(W_ih1, Wih1h, G3 * HH);
  cvt(W_hh1, Whh1h, G3 * HH);
  cvt(Wd1,   Wd1h,  DD1 * HH);
  cvt(Wd2,   Wd2h,  DD2 * DD1);

  ens_avg_kernel<<<(NTT * DD2 + 255) / 256, 256, 0, stream>>>(We, be, Wavg, bavg);

  // Layer 0: xi0 = emb[smiles] @ W_ih0^T + b_ih0  (gathered A)
  gemm_wmma_kernel<true, true><<<dim3(G3 / 256, MM / 32), 256, 0, stream>>>(
      nullptr, smiles, embh, Wih0h, b_ih0, xi, MM, G3, EE);
  gru_scan_kernel<<<16, 512, SCAN_LDS_TOTAL, stream>>>(xi, Whh0h, b_hh0, seq0, nullptr, 1);

  // Layer 1: xi1 = seq0 @ W_ih1^T + b_ih1  (reuses xi buffer)
  gemm_wmma_kernel<false, true><<<dim3(G3 / 256, MM / 32), 256, 0, stream>>>(
      seq0, nullptr, nullptr, Wih1h, b_ih1, xi, MM, G3, HH);
  gru_scan_kernel<<<16, 512, SCAN_LDS_TOTAL, stream>>>(xi, Whh1h, b_hh1, nullptr, hTh, 0);

  // Dense head
  gemm_wmma_kernel<false, false><<<dim3(DD1 / 256, BB / 32), 256, 0, stream>>>(
      hTh, nullptr, nullptr, Wd1h, bd1, d1out, BB, DD1, HH);
  bn_relu_kernel<<<(DD1 + 255) / 256, 256, 0, stream>>>(d1out, g1, beta1, h1h, BB, DD1);

  gemm_wmma_kernel<false, false><<<dim3(DD2 / 256, BB / 32), 256, 0, stream>>>(
      h1h, nullptr, nullptr, Wd2h, bd2, d2out, BB, DD2, DD1);
  bn_relu_kernel<<<(DD2 + 255) / 256, 256, 0, stream>>>(d2out, g2, beta2, h2h, BB, DD2);

  score_kernel<<<NTT, BB, 0, stream>>>(h2h, Wavg, bavg, out);

  (void)in_sizes; (void)n_in; (void)out_size; (void)ws_size;
}
